// BaseMAWAttention_12515534701255
// MI455X (gfx1250) — matrix operand order repaired and checked by reading, not compiled, not run on hardware
//
#include <hip/hip_runtime.h>
#include <hip/hip_bf16.h>

typedef __attribute__((ext_vector_type(16))) __bf16 v16bf;
typedef __attribute__((ext_vector_type(8)))  float  v8f;

#define HIDDEN   768
#define HEADS    12
#define HEAD_DIM 64
#define DEPTH    8
#define BATCH    4
#define SEQ      512
#define TOKENS   (BATCH * SEQ)

// ---------------------------------------------------------------------------
// WMMA fragment loaders (wave32, 16x16x32 bf16, per CDNA5 ISA 7.12.2)
// A operand: row-major (M x K), K contiguous.
//   lane l: M = l&15 ; VGPR0-3 = K[koff .. koff+7], VGPR4-7 = K[16+koff .. 23+koff]
//   where koff = (l&16) ? 8 : 0   -> two contiguous 16B loads.
// B operand: stored as B^T row-major (N x K), K contiguous.
//   lane l: N = l&15 ; VGPR0-3 = K[koff .. koff+7], VGPR4-7 = K[16+koff .. 23+koff]
//   where koff = (l&16) ? 8 : 0   -> two contiguous 16B loads (same as A).
// ---------------------------------------------------------------------------
__device__ __forceinline__ v16bf load_frag_a(const __bf16* __restrict__ base,
                                             int ld, int m0, int k0, int lane) {
    const __bf16* p = base + (size_t)(m0 + (lane & 15)) * ld + k0 + ((lane & 16) >> 1);
    union { uint4 u[2]; v16bf v; } r;
    r.u[0] = *(const uint4*)(p);
    r.u[1] = *(const uint4*)(p + 16);
    return r.v;
}

__device__ __forceinline__ v16bf load_frag_b(const __bf16* __restrict__ base,
                                             int ld, int n0, int k0, int lane) {
    const __bf16* p = base + (size_t)(n0 + (lane & 15)) * ld + k0 + ((lane & 16) >> 1);
    union { uint4 u[2]; v16bf v; } r;
    r.u[0] = *(const uint4*)(p);
    r.u[1] = *(const uint4*)(p + 16);
    return r.v;
}

__device__ __forceinline__ v8f wmma_bf16(v16bf a, v16bf b, v8f c) {
    return __builtin_amdgcn_wmma_f32_16x16x32_bf16(
        /*neg_a=*/false, a, /*neg_b=*/false, b,
        /*c_mod=*/(short)0, c, /*reuse_a=*/false, /*reuse_b=*/false);
}

// ---------------------------------------------------------------------------
// fp32 -> bf16 cast
// ---------------------------------------------------------------------------
__global__ void cast_bf16_kernel(const float* __restrict__ in,
                                 __bf16* __restrict__ out, int n) {
    int i = blockIdx.x * blockDim.x + threadIdx.x;
    if (i < n) out[i] = (__bf16)in[i];
}

// ---------------------------------------------------------------------------
// Projection GEMM:  P = X (TOKENS x HIDDEN) * W^T (N x HIDDEN, row-major) + bias
// 4 waves / block; all 4 waves share one 16-row A strip (block-uniform m-tile),
// staged ONCE into LDS with the CDNA5 async copy engine
// (global_load_async_to_lds_b128 + s_wait_asynccnt).
// Each wave computes a 16(M) x 64(N) tile: A fragment reused over 4 B tiles.
// mode 0: V  -> store transposed Vt[b][h][e][s]   (N = 768)
// mode 1: Qd/Kd -> store [b][h][d][s][e]          (N = 6144)
// ---------------------------------------------------------------------------
__global__ void __launch_bounds__(128)
proj_kernel(const __bf16* __restrict__ X, const __bf16* __restrict__ W,
            const float* __restrict__ bias, __bf16* __restrict__ out,
            int N, int mode) {
    __shared__ __bf16 s_a[16 * HIDDEN];                 // 24 KB A strip

    const int wave = threadIdx.x >> 5;
    const int lane = threadIdx.x & 31;
    const int ntiles64 = N >> 6;                        // 12 or 96: divisible by 4
    const int task  = blockIdx.x * 4 + wave;
    const int mtile = task / ntiles64;                  // block-uniform
    const int nt64  = task % ntiles64;
    const int m0 = mtile * 16;
    const int n0 = nt64 * 64;

    // ---- async stage X[m0 : m0+16, 0 : HIDDEN) -> LDS ----
    // 16*768*2 B = 24 KB = 1536 x 16B chunks; 128 threads x 12 issues,
    // each lane moves one 16B chunk per instruction.
    {
        const unsigned lds_base = (unsigned)(size_t)(void*)s_a;
        const unsigned long long gbase =
            (unsigned long long)(size_t)(X + (size_t)m0 * HIDDEN);
#pragma unroll
        for (int i = 0; i < 12; ++i) {
            const unsigned chunk = (unsigned)(i * 128 + threadIdx.x);
            const unsigned long long ga = gbase + (unsigned long long)chunk * 16ull;
            const unsigned la = lds_base + chunk * 16u;
            asm volatile("global_load_async_to_lds_b128 %0, %1, off"
                         :: "v"(la), "v"(ga) : "memory");
        }
        asm volatile("s_wait_asynccnt 0" ::: "memory");
    }
    __syncthreads();

    v8f acc[4] = {};
    for (int k0 = 0; k0 < HIDDEN; k0 += 32) {
        // keep the B stream warm in the cache hierarchy (global_prefetch_b8)
        if (k0 + 32 < HIDDEN)
            __builtin_prefetch(W + (size_t)(n0 + (lane & 15)) * HIDDEN + k0 + 32, 0, 1);
        v16bf a = load_frag_a(s_a, HIDDEN, 0, k0, lane);   // from LDS
#pragma unroll
        for (int t = 0; t < 4; ++t) {
            v16bf b = load_frag_b(W, HIDDEN, n0 + t * 16, k0, lane);
            acc[t] = wmma_bf16(a, b, acc[t]);
        }
    }

    const int mrow = (lane & 16) >> 1;   // 0 or 8
    const int ncol = lane & 15;
#pragma unroll
    for (int t = 0; t < 4; ++t) {
#pragma unroll
        for (int r = 0; r < 8; ++r) {
            const int m = m0 + r + mrow;
            const int n = n0 + t * 16 + ncol;
            const float v  = acc[t][r] + bias[n];
            const int  bb  = m / SEQ, s = m % SEQ;
            if (mode == 0) {
                const int h = n / HEAD_DIM, e = n % HEAD_DIM;
                out[((size_t)((bb * HEADS + h) * HEAD_DIM + e)) * SEQ + s] = (__bf16)v;
            } else {
                const int h = n / (HEAD_DIM * DEPTH);
                const int e = (n / DEPTH) % HEAD_DIM;
                const int d = n % DEPTH;
                out[(((size_t)((bb * HEADS + h) * DEPTH + d)) * SEQ + s) * HEAD_DIM + e] = (__bf16)v;
            }
        }
    }
}

// ---------------------------------------------------------------------------
// Attention per (b,h,d, 16-query block):
//   scores(16x512) = Q(16x64) * K^T  via WMMA  -> LDS (fp32)
//   row softmax (in-wave, shfl across half-row lane groups)
//   out(16x64)    = weights(16x512) * V  via WMMA (Vt stored N=e x K=s)
// One wave (32 threads) per block; 48 KB LDS.
// ---------------------------------------------------------------------------
__global__ void __launch_bounds__(32)
attn_kernel(const __bf16* __restrict__ Q,   // (B,H,D,S,E)
            const __bf16* __restrict__ K,   // (B,H,D,S,E)
            const __bf16* __restrict__ Vt,  // (B,H,E,S)
            const int*    __restrict__ mask,// (B,S)
            float*        __restrict__ out) // (D,B,H,S,E)
{
    __shared__ float  s_scores[16 * SEQ];    // 32 KB
    __shared__ __bf16 s_weights[16 * SEQ];   // 16 KB

    const int lane = threadIdx.x;
    int id = blockIdx.x;
    const int qb = id & 31; id >>= 5;
    const int d  = id & 7;  id >>= 3;
    const int h  = id % HEADS;
    const int b  = id / HEADS;

    const __bf16* Qm = Q  + ((size_t)((b * HEADS + h) * DEPTH + d)) * SEQ * HEAD_DIM;
    const __bf16* Km = K  + ((size_t)((b * HEADS + h) * DEPTH + d)) * SEQ * HEAD_DIM;
    const __bf16* Vm = Vt + ((size_t)(b * HEADS + h)) * HEAD_DIM * SEQ;
    const int q0 = qb * 16;

    const int mrow = (lane & 16) >> 1;
    const int ncol = lane & 15;

    // ---- scores = Q K^T / sqrt(64) ----
    v16bf a0 = load_frag_a(Qm, HEAD_DIM, q0, 0,  lane);
    v16bf a1 = load_frag_a(Qm, HEAD_DIM, q0, 32, lane);
    for (int nt = 0; nt < SEQ / 16; ++nt) {
        v8f c = {};
        v16bf b0 = load_frag_b(Km, HEAD_DIM, nt * 16, 0,  lane);
        c = wmma_bf16(a0, b0, c);
        v16bf b1 = load_frag_b(Km, HEAD_DIM, nt * 16, 32, lane);
        c = wmma_bf16(a1, b1, c);
#pragma unroll
        for (int r = 0; r < 8; ++r)
            s_scores[(r + mrow) * SEQ + nt * 16 + ncol] = c[r] * 0.125f;
    }
    __syncthreads();

    // ---- softmax over keys: lane handles row (lane&15), half (lane>>4) ----
    const int row = lane & 15;
    const int j0  = (lane >> 4) * (SEQ / 2);
    const int* mrow_mask = mask + b * SEQ;

    float mx = -1e30f;
    for (int j = 0; j < SEQ / 2; ++j) {
        float s = s_scores[row * SEQ + j0 + j];
        if (mrow_mask[j0 + j] == 0) s = -1.0e9f;
        mx = fmaxf(mx, s);
    }
    mx = fmaxf(mx, __shfl_xor(mx, 16, 32));

    float sum = 0.f;
    for (int j = 0; j < SEQ / 2; ++j) {
        float s = s_scores[row * SEQ + j0 + j];
        if (mrow_mask[j0 + j] == 0) s = -1.0e9f;
        sum += __expf(s - mx);
    }
    sum += __shfl_xor(sum, 16, 32);
    const float inv = 1.0f / sum;

    for (int j = 0; j < SEQ / 2; ++j) {
        float s = s_scores[row * SEQ + j0 + j];
        if (mrow_mask[j0 + j] == 0) s = -1.0e9f;
        s_weights[row * SEQ + j0 + j] = (__bf16)(__expf(s - mx) * inv);
    }
    __syncthreads();

    // ---- out = weights @ V ----
    v8f acc[4] = {};
    for (int k0 = 0; k0 < SEQ; k0 += 32) {
        v16bf a = load_frag_a(s_weights, SEQ, 0, k0, lane);
#pragma unroll
        for (int t = 0; t < 4; ++t) {
            v16bf bb = load_frag_b(Vm, SEQ, t * 16, k0, lane);
            acc[t] = wmma_bf16(a, bb, acc[t]);
        }
    }

    float* om = out + ((size_t)((d * BATCH + b) * HEADS + h)) * SEQ * HEAD_DIM;
#pragma unroll
    for (int t = 0; t < 4; ++t)
#pragma unroll
        for (int r = 0; r < 8; ++r)
            om[(size_t)(q0 + r + mrow) * HEAD_DIM + t * 16 + ncol] = acc[t][r];
}

// ---------------------------------------------------------------------------
// Launch
// ---------------------------------------------------------------------------
extern "C" void kernel_launch(void* const* d_in, const int* in_sizes, int n_in,
                              void* d_out, int out_size, void* d_ws, size_t ws_size,
                              hipStream_t stream) {
    const float* hidden = (const float*)d_in[0];
    const int*   amask  = (const int*)  d_in[1];
    const float* Wv     = (const float*)d_in[6];
    const float* bv     = (const float*)d_in[7];
    const float* Wqd    = (const float*)d_in[8];
    const float* bqd    = (const float*)d_in[9];
    const float* Wkd    = (const float*)d_in[10];
    const float* bkd    = (const float*)d_in[11];
    float* out = (float*)d_out;

    // workspace carve-up (bf16 buffers)
    char* ws = (char*)d_ws;
    __bf16* Xbf   = (__bf16*)ws;                     ws += (size_t)TOKENS * HIDDEN * 2;
    __bf16* Wvbf  = (__bf16*)ws;                     ws += (size_t)HIDDEN * HIDDEN * 2;
    __bf16* Wqdbf = (__bf16*)ws;                     ws += (size_t)HIDDEN * DEPTH * HIDDEN * 2;
    __bf16* Wkdbf = (__bf16*)ws;                     ws += (size_t)HIDDEN * DEPTH * HIDDEN * 2;
    __bf16* Qbuf  = (__bf16*)ws;                     ws += (size_t)BATCH * HEADS * DEPTH * SEQ * HEAD_DIM * 2;
    __bf16* Kbuf  = (__bf16*)ws;                     ws += (size_t)BATCH * HEADS * DEPTH * SEQ * HEAD_DIM * 2;
    __bf16* Vtbuf = (__bf16*)ws;

    // 1) casts to bf16
    {
        int n = TOKENS * HIDDEN;
        cast_bf16_kernel<<<(n + 255) / 256, 256, 0, stream>>>(hidden, Xbf, n);
        n = HIDDEN * HIDDEN;
        cast_bf16_kernel<<<(n + 255) / 256, 256, 0, stream>>>(Wv, Wvbf, n);
        n = HIDDEN * DEPTH * HIDDEN;
        cast_bf16_kernel<<<(n + 255) / 256, 256, 0, stream>>>(Wqd, Wqdbf, n);
        cast_bf16_kernel<<<(n + 255) / 256, 256, 0, stream>>>(Wkd, Wkdbf, n);
    }

    // 2) projections (each wave: 16x64 output tile; A strip async-staged in LDS)
    {
        int tasks = (TOKENS / 16) * (HIDDEN / 64);           // V: 128*12
        proj_kernel<<<(tasks + 3) / 4, 128, 0, stream>>>(Xbf, Wvbf, bv, Vtbuf, HIDDEN, 0);
        tasks = (TOKENS / 16) * ((HIDDEN * DEPTH) / 64);     // Qd/Kd: 128*96
        proj_kernel<<<(tasks + 3) / 4, 128, 0, stream>>>(Xbf, Wqdbf, bqd, Qbuf, HIDDEN * DEPTH, 1);
        proj_kernel<<<(tasks + 3) / 4, 128, 0, stream>>>(Xbf, Wkdbf, bkd, Kbuf, HIDDEN * DEPTH, 1);
    }

    // 3) attention: one wave per (b,h,d, 16-query block)
    {
        int blocks = BATCH * HEADS * DEPTH * (SEQ / 16);     // 12288
        attn_kernel<<<blocks, 32, 0, stream>>>(Qbuf, Kbuf, Vtbuf, amask, out);
    }
}
